// EnhancedMSAGAT_Net_25228637897512
// MI455X (gfx1250) — matrix operand order repaired and checked by reading, not compiled, not run on hardware
//
#include <hip/hip_runtime.h>

// ---------------- sizes ----------------
#define Bb 16
#define Ww 64
#define Nn 1024
#define Cc 128
#define Hh 8
#define BOT 16
#define HOR 24
#define Ss 5
#define Kk 3
#define HWW 16
#define HD 16
#define FD (Ss * Cc)          // 640 fused feature dim

typedef __bf16 v16bf __attribute__((ext_vector_type(16)));
typedef float  v8f   __attribute__((ext_vector_type(8)));

union Frag {
    v16bf v;
    unsigned short u[16];
    uint4 q[2];
};

__device__ __forceinline__ unsigned short f2bf(float f) {
    unsigned int x = __float_as_uint(f);
    unsigned int r = x + 0x7FFFu + ((x >> 16) & 1u);   // round-to-nearest-even
    return (unsigned short)(r >> 16);
}

// ============ Kernel W: fusion_w f32 -> bf16 ============
__global__ __launch_bounds__(256) void wconv_kernel(
    const float* __restrict__ w, unsigned short* __restrict__ wb)
{
    int i = blockIdx.x * 256 + threadIdx.x;      // grid sized exactly to Cc*FD
    wb[i] = f2bf(w[i]);
}

// ============ Kernel A1: dilated convs + ReLU-mean -> feats (bf16) ============
__global__ __launch_bounds__(128) void conv_kernel(
    const float* __restrict__ x, const float* __restrict__ conv_w,
    const float* __restrict__ conv_b, unsigned short* __restrict__ feats)
{
    __shared__ float xr[Ww];
    int bn = blockIdx.x;                // b*N + n
    int b = bn >> 10, n = bn & 1023;
    int c = threadIdx.x;                // 0..127 = output channel
    if (c < Ww) xr[c] = x[((size_t)b * Ww + c) * Nn + n];
    __syncthreads();
    // copy x-row into registers once; all conv taps become register reads
    float xv[Ww];
#pragma unroll
    for (int t = 0; t < Ww; ++t) xv[t] = xr[t];
#pragma unroll
    for (int s = 0; s < Ss; ++s) {
        const int d = 1 << s;           // dilation == padding
        float w0 = conv_w[(s * Cc + c) * Kk + 0];
        float w1 = conv_w[(s * Cc + c) * Kk + 1];
        float w2 = conv_w[(s * Cc + c) * Kk + 2];
        float bias = conv_b[s * Cc + c];
        float acc = 0.f;
#pragma unroll
        for (int t = 0; t < Ww; ++t) {
            float xm = (t - d >= 0) ? xv[t - d] : 0.f;
            float xp = (t + d < Ww) ? xv[t + d] : 0.f;
            float o = fmaf(w0, xm, fmaf(w1, xv[t], fmaf(w2, xp, bias)));
            acc += fmaxf(o, 0.f);
        }
        feats[(size_t)bn * FD + s * Cc + c] = f2bf(acc * (1.0f / (float)Ww));
    }
}

// ============ Kernel A2: fusion GEMM on WMMA: h = relu(feats @ W^T + b) ============
// feats: (B*N, 640) bf16 ; Wb: (128, 640) bf16 ; h: (B*N, 128) f32
#define GWAVES 8
__global__ __launch_bounds__(256) void fusion_gemm_kernel(
    const unsigned short* __restrict__ feats, const unsigned short* __restrict__ Wb,
    const float* __restrict__ fusion_b, float* __restrict__ hbuf)
{
    int wave = threadIdx.x >> 5;
    int lane = threadIdx.x & 31;
    int tile = blockIdx.x * GWAVES + wave;       // 0 .. 8191
    int ct = tile & 7;                           // 8 column tiles of 16
    int rt = tile >> 3;                          // 1024 row tiles of 16
    int half = lane >> 4;
    int ln   = lane & 15;
    int r0 = rt * 16, c0 = ct * 16;

    const unsigned short* arow = feats + (size_t)(r0 + ln) * FD;   // A: lane = row
    const unsigned short* brow = Wb    + (size_t)(c0 + ln) * FD;   // B: lane = col

    v8f acc = {};
#pragma unroll 4
    for (int kb = 0; kb < FD; kb += 32) {
        Frag a, bfr;
        // A fragment: a[0..7] = K[kb+8h ..], a[8..15] = K[kb+16+8h ..]
        a.q[0]  = *(const uint4*)(arow + kb + 8 * half);
        a.q[1]  = *(const uint4*)(arow + kb + 16 + 8 * half);
        // B fragment: b[0..7] = W[col][kb+8h ..], b[8..15] = W[col][kb+16+8h ..]
        bfr.q[0] = *(const uint4*)(brow + kb + 8 * half);
        bfr.q[1] = *(const uint4*)(brow + kb + 16 + 8 * half);
        acc = __builtin_amdgcn_wmma_f32_16x16x32_bf16(false, a.v, false, bfr.v, (short)0, acc, false, false);
    }
    float bias = fusion_b[c0 + ln];
#pragma unroll
    for (int r = 0; r < 8; ++r) {
        int row = r0 + r + 8 * half;
        hbuf[(size_t)row * Cc + c0 + ln] = fmaxf(acc[r] + bias, 0.f);
    }
}

// ============ Kernel B: LayerNorm + low-rank QKV + scatter to WMMA layouts ============
// Qext[b][h][n][0:16]=q*0.25, [16:32]=u[h][n][:]   (bf16, row stride 32)
// Kext[b][h][n][0:16]=k,      [16:32]=v[h][:][n]
// Vt  [b][h][d][n] = val[b][h][n][d]
__global__ __launch_bounds__(128) void qkv_kernel(
    const float* __restrict__ hbuf, const float* __restrict__ ln_g,
    const float* __restrict__ ln_b, const float* __restrict__ wql,
    const float* __restrict__ bql, const float* __restrict__ wqh,
    const float* __restrict__ bqh, const float* __restrict__ u,
    const float* __restrict__ v,
    unsigned short* __restrict__ Qext, unsigned short* __restrict__ Kext,
    unsigned short* __restrict__ Vt)
{
    __shared__ float hn[Cc];
    __shared__ float tmp[3 * BOT];
    __shared__ float red[Cc];
    int bn = blockIdx.x; int b = bn >> 10, n = bn & 1023;
    int c = threadIdx.x;
    float hv = hbuf[(size_t)bn * Cc + c];
    red[c] = hv; __syncthreads();
    for (int off = 64; off > 0; off >>= 1) { if (c < off) red[c] += red[c + off]; __syncthreads(); }
    float mu = red[0] * (1.f / Cc);
    __syncthreads();
    float dv = hv - mu;
    red[c] = dv * dv; __syncthreads();
    for (int off = 64; off > 0; off >>= 1) { if (c < off) red[c] += red[c + off]; __syncthreads(); }
    float var = red[0] * (1.f / Cc);
    hn[c] = dv * rsqrtf(var + 1e-5f) * ln_g[c] + ln_b[c];
    __syncthreads();
    if (c < 3 * BOT) {
        float a = bql[c];
        const float* wr = wql + c * Cc;
        for (int j = 0; j < Cc; ++j) a += hn[j] * wr[j];
        tmp[c] = a;
    }
    __syncthreads();
    int hh = c >> 4, d = c & 15;
    float q0 = bqh[c], q1 = bqh[Cc + c], q2 = bqh[2 * Cc + c];
    const float* w0 = wqh + (size_t)c * (3 * BOT);
    const float* w1 = wqh + (size_t)(Cc + c) * (3 * BOT);
    const float* w2 = wqh + (size_t)(2 * Cc + c) * (3 * BOT);
    for (int j = 0; j < 3 * BOT; ++j) { float t = tmp[j]; q0 += t * w0[j]; q1 += t * w1[j]; q2 += t * w2[j]; }
    size_t row = ((size_t)(b * Hh + hh) * Nn + n) * 32;
    Qext[row + d]      = f2bf(0.25f * q0);                                 // 1/sqrt(HD)
    Qext[row + 16 + d] = f2bf(u[((size_t)hh * Nn + n) * BOT + d]);
    Kext[row + d]      = f2bf(q1);
    Kext[row + 16 + d] = f2bf(v[((size_t)hh * BOT + d) * Nn + n]);
    Vt[((size_t)(b * Hh + hh) * HD + d) * Nn + n] = f2bf(q2);
}

// ============ Kernel C: flash attention on WMMA (S^T trick) ============
#define NWAVES 8
__global__ __launch_bounds__(256) void attn_kernel(
    const unsigned short* __restrict__ Qext, const unsigned short* __restrict__ Kext,
    const unsigned short* __restrict__ Vt, const int* __restrict__ adj,
    const float* __restrict__ adj_scale, float* __restrict__ ao)
{
    int wave = threadIdx.x >> 5;
    int lane = threadIdx.x & 31;
    int tile = blockIdx.x * NWAVES + wave;      // 0..8191
    int rb = tile & 63;
    int h  = (tile >> 6) & 7;
    int b  = tile >> 9;
    int half = lane >> 4;
    int ln   = lane & 15;
    int r0 = rb * 16;

    const float NEGM = -1.0e9f * adj_scale[0];
    size_t bhBase = (size_t)(b * Hh + h) * Nn;

    // B-fragment of Qext^T: lane = query column, chunks K[8h..], K[16+8h..]
    Frag bq;
    {
        const uint4* p = (const uint4*)(Qext + (bhBase + r0 + ln) * 32);
        bq.q[0] = p[half]; bq.q[1] = p[2 + half];
    }

    v8f oacc = {};
    float m_run = -3.0e38f;
    float l_run = 0.f;
    const int* adjrow = adj + (size_t)(r0 + ln) * Nn;
    const unsigned short* vrow = Vt + ((size_t)(b * Hh + h) * HD + ln) * Nn;

    for (int it = 0; it < Nn / 32; ++it) {
        int kb = it * 32;
        // A-fragments: Kext rows (keys), documented 16-bit A chunking
        Frag aK0, aK1;
        {
            const uint4* p0 = (const uint4*)(Kext + (bhBase + kb + ln) * 32);
            aK0.q[0] = p0[half];     aK0.q[1] = p0[2 + half];
            const uint4* p1 = (const uint4*)(Kext + (bhBase + kb + 16 + ln) * 32);
            aK1.q[0] = p1[half];     aK1.q[1] = p1[2 + half];
        }
        if (it + 1 < Nn / 32) {     // uniform branch: prefetch next K/V tiles
            __builtin_prefetch(Kext + (bhBase + kb + 32 + ln) * 32, 0, 3);
            __builtin_prefetch(vrow + kb + 32 + 16 * half, 0, 3);
        }
        v8f z = {};
        v8f s0 = __builtin_amdgcn_wmma_f32_16x16x32_bf16(false, aK0.v, false, bq.v, (short)0, z, false, false);
        v8f s1 = __builtin_amdgcn_wmma_f32_16x16x32_bf16(false, aK1.v, false, bq.v, (short)0, z, false, false);

        // mask + tile max; s0[r]: key = kb + r + 8*half, query = r0 + ln
        const int* am0 = adjrow + kb + 8 * half;
        const int* am1 = am0 + 16;
        float tm = -3.0e38f;
#pragma unroll
        for (int r = 0; r < 8; ++r) {
            s0[r] += (am0[r] > 0) ? 0.f : NEGM;
            s1[r] += (am1[r] > 0) ? 0.f : NEGM;
            tm = fmaxf(tm, fmaxf(s0[r], s1[r]));
        }
        tm = fmaxf(tm, __shfl_xor(tm, 16, 32));
        float m_new = fmaxf(m_run, tm);
        float alpha = __expf(m_run - m_new);
        float ts = 0.f;
        Frag pa;   // P tile as bf16 A-fragment (no cross-lane movement needed)
#pragma unroll
        for (int r = 0; r < 8; ++r) {
            float p0 = __expf(s0[r] - m_new);
            float p1 = __expf(s1[r] - m_new);
            ts += p0 + p1;
            pa.u[r]     = f2bf(p0);
            pa.u[8 + r] = f2bf(p1);
        }
        ts += __shfl_xor(ts, 16, 32);
        l_run = l_run * alpha + ts;
        m_run = m_new;

        // V B-fragment: lane = d column, chunks K[8h..], K[16+8h..] from Vt
        Frag bv;
        {
            const uint4* p = (const uint4*)(vrow + kb);
            bv.q[0] = p[half]; bv.q[1] = p[2 + half];
        }
        // rescale running output (per-query alpha lives in lane space -> bpermute)
#pragma unroll
        for (int r = 0; r < 8; ++r) {
            float ar = __shfl(alpha, r + 8 * half, 32);
            oacc[r] *= ar;
        }
        oacc = __builtin_amdgcn_wmma_f32_16x16x32_bf16(false, pa.v, false, bv.v, (short)0, oacc, false, false);
    }
    float linv = 1.0f / l_run;
#pragma unroll
    for (int r = 0; r < 8; ++r) {
        float li = __shfl(linv, r + 8 * half, 32);
        int n = r0 + r + 8 * half;
        ao[((size_t)b * Nn + n) * Cc + h * HD + ln] = oacc[r] * li;
    }
}

// ============ Kernel D: output projection + residual + predictor + horizon ============
__global__ __launch_bounds__(128) void head_kernel(
    const float* __restrict__ ao, const float* __restrict__ hbuf,
    const float* __restrict__ x,
    const float* __restrict__ wol, const float* __restrict__ bol,
    const float* __restrict__ woh, const float* __restrict__ boh,
    const float* __restrict__ p1_w, const float* __restrict__ p1_b,
    const float* __restrict__ p2_w, const float* __restrict__ p2_b,
    const float* __restrict__ hw_w, const float* __restrict__ hw_b,
    float* __restrict__ out)
{
    __shared__ float av[Cc];
    __shared__ float t1[BOT];
    __shared__ float ov[Cc];
    __shared__ float pf[BOT];
    int bn = blockIdx.x; int b = bn >> 10, n = bn & 1023;
    int c = threadIdx.x;
    av[c] = ao[(size_t)bn * Cc + c];
    __syncthreads();
    if (c < BOT) {
        float a = bol[c]; const float* wr = wol + c * Cc;
        for (int j = 0; j < Cc; ++j) a += av[j] * wr[j];
        t1[c] = a;
    }
    __syncthreads();
    {
        float a = boh[c] + hbuf[(size_t)bn * Cc + c];   // + residual
        const float* wr = woh + c * BOT;
        for (int j = 0; j < BOT; ++j) a += t1[j] * wr[j];
        ov[c] = a;
    }
    __syncthreads();
    if (c < BOT) {
        float a = p1_b[c]; const float* wr = p1_w + c * Cc;
        for (int j = 0; j < Cc; ++j) a += ov[j] * wr[j];
        pf[c] = fmaxf(a, 0.f);
    }
    __syncthreads();
    if (c < HOR) {
        float a = p2_b[c] + hw_b[c];
        const float* wr = p2_w + c * BOT;
        for (int j = 0; j < BOT; ++j) a += pf[j] * wr[j];
        const float* hr = hw_w + c * HWW;
        for (int w = 0; w < HWW; ++w)
            a += x[((size_t)b * Ww + (Ww - HWW) + w) * Nn + n] * hr[w];
        out[((size_t)b * HOR + c) * Nn + n] = a;
    }
    if (bn == 0 && c == 0) out[(size_t)Bb * HOR * Nn] = 0.0f;   // attn_reg_loss
}

// ============ launch ============
extern "C" void kernel_launch(void* const* d_in, const int* in_sizes, int n_in,
                              void* d_out, int out_size, void* d_ws, size_t ws_size,
                              hipStream_t stream)
{
    const float* x        = (const float*)d_in[0];
    const int*   adj      = (const int*)  d_in[1];
    const float* conv_w   = (const float*)d_in[2];
    const float* conv_b   = (const float*)d_in[3];
    const float* fusion_w = (const float*)d_in[4];
    const float* fusion_b = (const float*)d_in[5];
    const float* ln_g     = (const float*)d_in[6];
    const float* ln_b     = (const float*)d_in[7];
    const float* wql      = (const float*)d_in[8];
    const float* bql      = (const float*)d_in[9];
    const float* wqh      = (const float*)d_in[10];
    const float* bqh      = (const float*)d_in[11];
    const float* u        = (const float*)d_in[12];
    const float* v        = (const float*)d_in[13];
    const float* adjs     = (const float*)d_in[14];
    const float* wol      = (const float*)d_in[15];
    const float* bol      = (const float*)d_in[16];
    const float* woh      = (const float*)d_in[17];
    const float* boh      = (const float*)d_in[18];
    const float* p1_w     = (const float*)d_in[19];
    const float* p1_b     = (const float*)d_in[20];
    const float* p2_w     = (const float*)d_in[21];
    const float* p2_b     = (const float*)d_in[22];
    const float* hw_w     = (const float*)d_in[23];
    const float* hw_b     = (const float*)d_in[24];

    char* ws = (char*)d_ws;
    const size_t SZ_H  = (size_t)Bb * Nn * Cc * sizeof(float);          // 8 MB
    const size_t SZ_QE = (size_t)Bb * Hh * Nn * 32 * sizeof(short);     // 8 MB
    const size_t SZ_VT = (size_t)Bb * Hh * HD * Nn * sizeof(short);     // 4 MB
    const size_t SZ_FT = (size_t)Bb * Nn * FD * sizeof(short);          // 20 MB
    float*          hbuf  = (float*)(ws);
    float*          ao    = (float*)(ws + SZ_H);
    unsigned short* Qext  = (unsigned short*)(ws + 2 * SZ_H);
    unsigned short* Kext  = (unsigned short*)(ws + 2 * SZ_H + SZ_QE);
    unsigned short* Vt    = (unsigned short*)(ws + 2 * SZ_H + 2 * SZ_QE);
    unsigned short* featb = (unsigned short*)(ws + 2 * SZ_H + 2 * SZ_QE + SZ_VT);
    unsigned short* Wb    = (unsigned short*)(ws + 2 * SZ_H + 2 * SZ_QE + SZ_VT + SZ_FT);

    wconv_kernel<<<(Cc * FD) / 256, 256, 0, stream>>>(fusion_w, Wb);
    conv_kernel<<<Bb * Nn, Cc, 0, stream>>>(x, conv_w, conv_b, featb);
    fusion_gemm_kernel<<<((Bb * Nn / 16) * (Cc / 16)) / GWAVES, 32 * GWAVES, 0, stream>>>(featb, Wb, fusion_b, hbuf);
    qkv_kernel<<<Bb * Nn, Cc, 0, stream>>>(hbuf, ln_g, ln_b, wql, bql, wqh, bqh, u, v, Qext, Kext, Vt);
    attn_kernel<<<(Bb * Hh * (Nn / 16)) / NWAVES, 32 * NWAVES, 0, stream>>>(Qext, Kext, Vt, adj, adjs, ao);
    head_kernel<<<Bb * Nn, Cc, 0, stream>>>(ao, hbuf, x, wol, bol, woh, boh,
                                            p1_w, p1_b, p2_w, p2_b, hw_w, hw_b, (float*)d_out);
}
